// SparseAttention_31937376813306
// MI455X (gfx1250) — matrix operand order repaired and checked by reading, not compiled, not run on hardware
//
#include <hip/hip_runtime.h>
#include <hip/hip_bf16.h>
#include <hip/hip_fp16.h>

// ---------------- problem constants ----------------
#define NQ   4096      // sequence length
#define DIMC 512       // model dim
#define HEADS 8
#define HD   64        // head dim (KQ == VD == 64)
#define TOPK 32

typedef __attribute__((ext_vector_type(16))) __bf16 v16bf;
typedef __attribute__((ext_vector_type(8)))  float  v8f;

union Frag16 { v16bf v; unsigned short u[16]; uint4 q[2]; };

// ---------------- helpers ----------------
__device__ __forceinline__ unsigned short f2bf(float f) {
    unsigned int u = __float_as_uint(f);
    unsigned int r = u + 0x7FFFu + ((u >> 16) & 1u);   // RNE
    return (unsigned short)(r >> 16);
}
__device__ __forceinline__ unsigned int f32_sortable(float f) {
    unsigned int u = __float_as_uint(f);
    return u ^ ((unsigned int)(((int)u) >> 31) | 0x80000000u);
}
__device__ __forceinline__ float sortable_f32(unsigned int s) {
    unsigned int u = s ^ ((unsigned int)(((int)(~s)) >> 31) | 0x80000000u);
    return __uint_as_float(u);
}
__device__ __forceinline__ unsigned long long shfl_xor_u64(unsigned long long v, int off) {
    int lo = __shfl_xor((int)(unsigned int)v, off, 32);
    int hi = __shfl_xor((int)(unsigned int)(v >> 32), off, 32);
    return ((unsigned long long)(unsigned int)hi << 32) | (unsigned int)lo;
}

// ---------------- kernel 0a: fp32 -> bf16 (row-major copy, for A matrices) ---
__global__ void f32_to_bf16_kernel(const float* __restrict__ src,
                                   unsigned short* __restrict__ dst, int n) {
    int i = blockIdx.x * blockDim.x + threadIdx.x;
    if (i < n) dst[i] = f2bf(src[i]);
}

// ---------------- kernel 0b: fp32 [K][N] -> bf16 transposed [N][K] ------------
__global__ void f32_to_bf16_t_kernel(const float* __restrict__ src,
                                     unsigned short* __restrict__ dst,
                                     int K, int N) {
    int i = blockIdx.x * blockDim.x + threadIdx.x;   // i = n*K + k (write-coalesced)
    if (i < K * N) {
        int n = i / K, k = i % K;
        dst[i] = f2bf(src[(size_t)k * N + n]);
    }
}

// ---------------- kernel 1/4: WMMA GEMM  C = A(bf16 MxK) @ Bt(bf16 [N][K])^T + bias
// MODE 0: fp32 row-major [M][N]
// MODE 1: bf16 head-major out[(h*M + m)*64 + d], h = n>>6, d = n&63
// MODE 2: fp32 head-major (same mapping)
template <int MODE>
__global__ __launch_bounds__(128)
void wmma_gemm_bf16_kernel(const unsigned short* __restrict__ A,
                           const unsigned short* __restrict__ Bt,
                           const float* __restrict__ bias,
                           float* __restrict__ outF,
                           unsigned short* __restrict__ outB,
                           int M, int N, int Kd) {
    const int lane = threadIdx.x & 31;
    const int wave = threadIdx.x >> 5;
    const int tileM = blockIdx.x * 64 + wave * 16;
    const int tileN = blockIdx.y * 64;

    const int ah = (lane >> 4) << 4;   // A half-wave K offset (16h)
    const int bh = (lane >> 4) << 4;   // B half-wave K offset (16h)
    const int bc = lane & 15;          // B column within 16-tile

    const unsigned short* Arow = A + (size_t)(tileM + (lane & 15)) * Kd;
    const unsigned short* Brow0 = Bt + (size_t)(tileN + 0  + bc) * Kd + bh;
    const unsigned short* Brow1 = Bt + (size_t)(tileN + 16 + bc) * Kd + bh;
    const unsigned short* Brow2 = Bt + (size_t)(tileN + 32 + bc) * Kd + bh;
    const unsigned short* Brow3 = Bt + (size_t)(tileN + 48 + bc) * Kd + bh;

    v8f acc[4];
#pragma unroll
    for (int t = 0; t < 4; ++t)
        acc[t] = (v8f){0.f, 0.f, 0.f, 0.f, 0.f, 0.f, 0.f, 0.f};

#pragma unroll 2
    for (int kb = 0; kb < Kd; kb += 32) {
        __builtin_prefetch(Arow + kb + 64, 0, 1);    // global_prefetch_b8
        Frag16 a;
        a.q[0] = *(const uint4*)(Arow + kb + ah);        // K = kb+16h .. +7
        a.q[1] = *(const uint4*)(Arow + kb + 8 + ah);    // K = kb+16h+8 .. +7

        Frag16 b0, b1, b2, b3;
        b0.q[0] = *(const uint4*)(Brow0 + kb);  b0.q[1] = *(const uint4*)(Brow0 + kb + 8);
        b1.q[0] = *(const uint4*)(Brow1 + kb);  b1.q[1] = *(const uint4*)(Brow1 + kb + 8);
        b2.q[0] = *(const uint4*)(Brow2 + kb);  b2.q[1] = *(const uint4*)(Brow2 + kb + 8);
        b3.q[0] = *(const uint4*)(Brow3 + kb);  b3.q[1] = *(const uint4*)(Brow3 + kb + 8);

        acc[0] = __builtin_amdgcn_wmma_f32_16x16x32_bf16(false, a.v, false, b0.v,
                                                         (short)0, acc[0], false, false);
        acc[1] = __builtin_amdgcn_wmma_f32_16x16x32_bf16(false, a.v, false, b1.v,
                                                         (short)0, acc[1], false, false);
        acc[2] = __builtin_amdgcn_wmma_f32_16x16x32_bf16(false, a.v, false, b2.v,
                                                         (short)0, acc[2], false, false);
        acc[3] = __builtin_amdgcn_wmma_f32_16x16x32_bf16(false, a.v, false, b3.v,
                                                         (short)0, acc[3], false, false);
    }

#pragma unroll
    for (int j = 0; j < 8; ++j) {
        const int row = tileM + j + ((lane >> 4) << 3);
#pragma unroll
        for (int nb = 0; nb < 4; ++nb) {
            const int col = tileN + nb * 16 + bc;
            const float val = acc[nb][j] + bias[col];
            if (MODE == 0) {
                outF[(size_t)row * N + col] = val;
            } else {
                const int h = col >> 6, d = col & 63;
                const size_t o = ((size_t)h * M + row) * 64 + d;
                if (MODE == 1) outB[o] = f2bf(val);
                else           outF[o] = val;
            }
        }
    }
}

// ---------------- kernel 2: scores (WMMA) + top-32 selection ----------------
// grid (NQ/16, HEADS), block 256 (8 waves)
// dyn LDS: fp16 scores [16][4096] (128KB) + u64 lists [8 waves][32 ent][32 lanes] (64KB)
__global__ __launch_bounds__(256)
void scores_topk_kernel(const unsigned short* __restrict__ Qb,   // [H][NQ][64] bf16
                        const unsigned short* __restrict__ Kb,   // [H][NQ][64] bf16
                        int*   __restrict__ topk_idx,            // [H][NQ][32]
                        float* __restrict__ topk_score) {        // [H][NQ][32]
    extern __shared__ char smem_raw[];
    __half* S = reinterpret_cast<__half*>(smem_raw);                         // 16*4096
    unsigned long long* lists =
        reinterpret_cast<unsigned long long*>(smem_raw + 16 * NQ * 2);       // 8*32*32

    const int lane = threadIdx.x & 31;
    const int wave = threadIdx.x >> 5;
    const int h  = blockIdx.y;
    const int mq = blockIdx.x * 16;

    const unsigned short* Qh = Qb + (size_t)h * NQ * 64;
    const unsigned short* Kh = Kb + (size_t)h * NQ * 64;

    const int ah = (lane >> 4) << 4;
    const int bh = (lane >> 4) << 4;
    const int bc = lane & 15;

    // ---- phase 1: S[0:16][0:4096] = (Q_tile @ K^T) * 1/sqrt(64)
    Frag16 a0, a1;
    {
        const unsigned short* Qrow = Qh + (size_t)(mq + (lane & 15)) * 64;
        a0.q[0] = *(const uint4*)(Qrow + ah);
        a0.q[1] = *(const uint4*)(Qrow + 8 + ah);
        a1.q[0] = *(const uint4*)(Qrow + 32 + ah);
        a1.q[1] = *(const uint4*)(Qrow + 40 + ah);
    }
#pragma unroll 2
    for (int t = wave; t < NQ / 16; t += 8) {
        const int nB = t * 16;
        const unsigned short* Krow = Kh + (size_t)(nB + bc) * 64 + bh;
        Frag16 b0, b1;
        b0.q[0] = *(const uint4*)(Krow);       b0.q[1] = *(const uint4*)(Krow + 8);
        b1.q[0] = *(const uint4*)(Krow + 32);  b1.q[1] = *(const uint4*)(Krow + 40);

        v8f acc = (v8f){0.f, 0.f, 0.f, 0.f, 0.f, 0.f, 0.f, 0.f};
        acc = __builtin_amdgcn_wmma_f32_16x16x32_bf16(false, a0.v, false, b0.v,
                                                      (short)0, acc, false, false);
        acc = __builtin_amdgcn_wmma_f32_16x16x32_bf16(false, a1.v, false, b1.v,
                                                      (short)0, acc, false, false);
#pragma unroll
        for (int j = 0; j < 8; ++j) {
            const int m = j + ((lane >> 4) << 3);
            S[m * NQ + nB + bc] = __float2half(acc[j] * 0.125f);
        }
    }
    __syncthreads();

    // ---- phase 2: per-row top-32 (wave w handles rows 2w, 2w+1)
    const int listBase = wave * 32 * 32 + lane;   // entry i at listBase + i*32
    for (int r = 0; r < 2; ++r) {
        const int m = 2 * wave + r;
        const int q = mq + m;

        // init candidate list (sentinel 0 < any sortable float)
#pragma unroll
        for (int i = 0; i < 32; ++i) lists[listBase + i * 32] = 0ull;

        // lane scans its 128 keys, keeps sorted (desc) top-32 in LDS
        for (int c = lane; c < NQ; c += 32) {
            const float f = __half2float(S[m * NQ + c]);
            const unsigned long long key =
                ((unsigned long long)f32_sortable(f) << 32) | (unsigned int)c;
            if (key > lists[listBase + 31 * 32]) {
                int pos = 31;
                while (pos > 0 && lists[listBase + (pos - 1) * 32] < key) {
                    lists[listBase + pos * 32] = lists[listBase + (pos - 1) * 32];
                    --pos;
                }
                lists[listBase + pos * 32] = key;
            }
        }

        // wave-parallel 32-round merge of 32 sorted streams
        int p = 0;
        for (int rr = 0; rr < TOPK; ++rr) {
            unsigned long long cand = (p < 32) ? lists[listBase + p * 32] : 0ull;
            unsigned long long mx = cand;
#pragma unroll
            for (int off = 16; off; off >>= 1) {
                unsigned long long o = shfl_xor_u64(mx, off);
                if (o > mx) mx = o;
            }
            unsigned long long mask = __ballot(cand == mx);
            int src = (int)__ffsll(mask) - 1;
            if (lane == src) ++p;
            if (lane == rr) {
                const size_t o = ((size_t)h * NQ + q) * TOPK + rr;
                topk_idx[o]   = (int)(unsigned int)(mx & 0xFFFFFFFFull);
                topk_score[o] = sortable_f32((unsigned int)(mx >> 32));
            }
        }
    }
}

// ---------------- kernel 3: softmax over 32 neighbors + weighted V gather ------
// one wave per (h, n); block 256 => 8 (h,n) pairs per block
__global__ __launch_bounds__(256)
void attn_out_kernel(const float* __restrict__ V,          // [H][NQ][64] fp32
                     const int*   __restrict__ topk_idx,
                     const float* __restrict__ topk_score,
                     unsigned short* __restrict__ attnb) { // [NQ][512] bf16
    const int lane = threadIdx.x & 31;
    const int wave = threadIdx.x >> 5;
    const int gid  = blockIdx.x * 8 + wave;     // 0 .. H*NQ-1
    const int h = gid / NQ;
    const int n = gid % NQ;

    const size_t base = ((size_t)h * NQ + n) * TOPK + lane;
    float s  = topk_score[base];
    int   ix = topk_idx[base];

    float mval = s;
#pragma unroll
    for (int off = 16; off; off >>= 1) mval = fmaxf(mval, __shfl_xor(mval, off, 32));
    float e = __expf(s - mval);
    float sum = e;
#pragma unroll
    for (int off = 16; off; off >>= 1) sum += __shfl_xor(sum, off, 32);
    const float w = e / sum;

    float acc0 = 0.f, acc1 = 0.f;
    const float* Vh = V + (size_t)h * NQ * 64;
#pragma unroll 4
    for (int k = 0; k < TOPK; ++k) {
        const float wk = __shfl(w, k, 32);
        const int   ik = __shfl(ix, k, 32);
        const float* vp = Vh + (size_t)ik * 64;
        acc0 += wk * vp[lane];
        acc1 += wk * vp[lane + 32];
    }
    attnb[(size_t)n * DIMC + h * 64 + lane]      = f2bf(acc0);
    attnb[(size_t)n * DIMC + h * 64 + lane + 32] = f2bf(acc1);
}

// ---------------- host-side launch ----------------
extern "C" void kernel_launch(void* const* d_in, const int* in_sizes, int n_in,
                              void* d_out, int out_size, void* d_ws, size_t ws_size,
                              hipStream_t stream) {
    const float* x  = (const float*)d_in[0];
    const float* Wq = (const float*)d_in[1];
    const float* bq = (const float*)d_in[2];
    const float* Wk = (const float*)d_in[3];
    const float* bk = (const float*)d_in[4];
    const float* Wv = (const float*)d_in[5];
    const float* bv = (const float*)d_in[6];
    const float* Wo = (const float*)d_in[7];
    const float* bo = (const float*)d_in[8];
    float* out = (float*)d_out;

    // workspace carve-up (256B aligned)
    size_t off = 0;
    auto carve = [&](size_t bytes) -> char* {
        char* p = (char*)d_ws + off;
        off += (bytes + 255) & ~(size_t)255;
        return p;
    };
    unsigned short* xb    = (unsigned short*)carve((size_t)NQ * DIMC * 2);
    unsigned short* Wqt   = (unsigned short*)carve((size_t)DIMC * DIMC * 2);  // [N][K]
    unsigned short* Wkt   = (unsigned short*)carve((size_t)DIMC * DIMC * 2);
    unsigned short* Wvt   = (unsigned short*)carve((size_t)DIMC * DIMC * 2);
    unsigned short* Wot   = (unsigned short*)carve((size_t)DIMC * DIMC * 2);
    unsigned short* Qb    = (unsigned short*)carve((size_t)HEADS * NQ * HD * 2);
    unsigned short* Kbm   = (unsigned short*)carve((size_t)HEADS * NQ * HD * 2);
    float*          Vm    = (float*)carve((size_t)HEADS * NQ * HD * 4);
    int*            tidx  = (int*)carve((size_t)HEADS * NQ * TOPK * 4);
    float*          tsc   = (float*)carve((size_t)HEADS * NQ * TOPK * 4);
    unsigned short* attnb = (unsigned short*)carve((size_t)NQ * DIMC * 2);
    (void)ws_size; (void)n_in; (void)in_sizes; (void)out_size;

    // 0) conversions (weights transposed to [N][K] for contiguous B-fragments)
    {
        int n = NQ * DIMC;
        f32_to_bf16_kernel<<<(n + 255) / 256, 256, 0, stream>>>(x, xb, n);
        n = DIMC * DIMC;
        f32_to_bf16_t_kernel<<<(n + 255) / 256, 256, 0, stream>>>(Wq, Wqt, DIMC, DIMC);
        f32_to_bf16_t_kernel<<<(n + 255) / 256, 256, 0, stream>>>(Wk, Wkt, DIMC, DIMC);
        f32_to_bf16_t_kernel<<<(n + 255) / 256, 256, 0, stream>>>(Wv, Wvt, DIMC, DIMC);
        f32_to_bf16_t_kernel<<<(n + 255) / 256, 256, 0, stream>>>(Wo, Wot, DIMC, DIMC);
    }

    // 1) QKV projections (WMMA)
    {
        dim3 grid(NQ / 64, DIMC / 64), block(128);
        wmma_gemm_bf16_kernel<1><<<grid, block, 0, stream>>>(
            xb, Wqt, bq, nullptr, Qb, NQ, DIMC, DIMC);
        wmma_gemm_bf16_kernel<1><<<grid, block, 0, stream>>>(
            xb, Wkt, bk, nullptr, Kbm, NQ, DIMC, DIMC);
        wmma_gemm_bf16_kernel<2><<<grid, block, 0, stream>>>(
            xb, Wvt, bv, Vm, nullptr, NQ, DIMC, DIMC);
    }

    // 2) scores + top-K (LDS-resident 16x4096 fp16 slab + selection lists)
    {
        dim3 grid(NQ / 16, HEADS), block(256);
        size_t smem = (size_t)16 * NQ * 2                    // fp16 scores: 128KB
                    + (size_t)8 * 32 * 32 * 8;               // u64 lists:   64KB
        scores_topk_kernel<<<grid, block, smem, stream>>>(Qb, Kbm, tidx, tsc);
    }

    // 3) local softmax attention
    {
        dim3 grid(HEADS * NQ / 8), block(256);
        attn_out_kernel<<<grid, block, 0, stream>>>(Vm, tidx, tsc, attnb);
    }

    // 4) output projection (WMMA) -> fp32 d_out
    {
        dim3 grid(NQ / 64, DIMC / 64), block(128);
        wmma_gemm_bf16_kernel<0><<<grid, block, 0, stream>>>(
            attnb, Wot, bo, out, nullptr, NQ, DIMC, DIMC);
    }
}
